// IntraRNN_31602369364327
// MI455X (gfx1250) — compile-verified, matrix operand order checked by reading, not run on hardware
//
#include <hip/hip_runtime.h>
#include <hip/hip_bf16.h>

// ---------------------------------------------------------------------------
// Shapes
#define B_  1024
#define M_  50
#define H_  128
#define E_  128
#define NI_ 100000

// d_out flat layout (tuple return, concatenated):
#define OUT0 0LL                      // output      (1024,1,100000)
#define OUT1 102400000LL              // h_new       (1,1024,128)
#define OUT2 102531072LL              // embedded_in (1024,1,128)
#define OUT3 102662144LL              // gru_output  (1024,1,128)
#define OUT4 102793216LL              // attn_w      (1024,50)

typedef __attribute__((ext_vector_type(16))) _Float16 v16h;
typedef __attribute__((ext_vector_type(8)))  _Float16 v8h;
typedef __attribute__((ext_vector_type(8)))  float    v8f;

// ---------------------------------------------------------------------------
// Fast transcendentals. gfx1250 has V_TANH_F32 (TRANS op); use the builtin if
// this clang exposes it, else a branch-free __expf-based form (no libm branches
// next to the WMMA loop).
__device__ __forceinline__ float fast_tanh(float x) {
#if __has_builtin(__builtin_amdgcn_tanhf)
  return __builtin_amdgcn_tanhf(x);
#else
  float e = __expf(2.f * x);
  return 1.f - 2.f / (e + 1.f);
#endif
}

__device__ __forceinline__ float fast_sigmoid(float x) {
  return 1.f / (1.f + __expf(-x));
}

// ---------------------------------------------------------------------------
// WMMA fragment loaders (CDNA5 16x16x32 f16 layout, wave32).
// A/B operand: lane holds row (lane&15); K split: lanes 0-15 take k=[k0..k0+7]
// and [k0+16..k0+23], lanes 16-31 take [k0+8..k0+15] and [k0+24..k0+31].
__device__ __forceinline__ v16h load_frag_lds(const _Float16* base, int row0,
                                              int ld, int k0, int lane) {
  const _Float16* p = base + (row0 + (lane & 15)) * ld + k0 + ((lane & 16) ? 8 : 0);
  v8h lo = *(const v8h*)(p);
  v8h hi = *(const v8h*)(p + 16);
  v16h r;
#pragma unroll
  for (int e = 0; e < 8; ++e) { r[e] = lo[e]; r[8 + e] = hi[e]; }
  return r;
}

__device__ __forceinline__ v16h load_frag_gf32(const float* base, int row0,
                                               int ld, int k0, int lane) {
  const float* p = base + (size_t)(row0 + (lane & 15)) * ld + k0 + ((lane & 16) ? 8 : 0);
  v16h r;
#pragma unroll
  for (int e = 0; e < 8; ++e) {
    r[e]     = (_Float16)p[e];
    r[8 + e] = (_Float16)p[e + 16];
  }
  return r;
}

// ---------------------------------------------------------------------------
// Kernel A: per-batch attention.
//   result = tanh(cat_in @ Wc^T + bc)  (WMMA f16->f32)
//   energies = result @ Ws^T + bs ; attn = softmax_m ; context = attn @ inter
// One block (128 thr = 4 waves) per batch element. cat_W (131 MB) is
// L2-resident (192 MB L2), read exactly once per block.
__global__ __launch_bounds__(128) void attn_kernel(
    const float* __restrict__ hidden, const float* __restrict__ inter,
    const int* __restrict__ users, const float* __restrict__ cat_W,
    const float* __restrict__ cat_b, const float* __restrict__ scale_W,
    const float* __restrict__ scale_b, float* __restrict__ attn_out,
    float* __restrict__ ctx_out) {
  __shared__ _Float16 sCat[64 * 256];   // 32 KB: [hidden | inter], M padded to 64
  __shared__ float sEnergy[64];
  __shared__ float sAttn[64];

  const int b = blockIdx.x;
  const int tid = threadIdx.x;
  const int user = users[b];

  if (tid < 64) sEnergy[tid] = 0.f;

  const float* hid = hidden + (size_t)b * H_;
  const float* itb = inter + (size_t)b * M_ * H_;
  for (int idx = tid; idx < 64 * 256; idx += 128) {
    int m = idx >> 8, c = idx & 255;
    float v = (c < 128) ? hid[c] : ((m < M_) ? itb[m * H_ + (c - 128)] : 0.f);
    sCat[idx] = (_Float16)v;
  }
  __syncthreads();

  const int wave = tid >> 5, lane = tid & 31;
  const float* Wc = cat_W + (size_t)user * H_ * 256;
  const float* bc = cat_b + (size_t)user * H_;
  const float* Ws = scale_W + (size_t)user * H_;

#pragma unroll
  for (int ni = 0; ni < 2; ++ni) {
    const int nt = wave + ni * 4;            // h-tile 0..7
    v8f acc[4] = {v8f{}, v8f{}, v8f{}, v8f{}};
    for (int kc = 0; kc < 8; ++kc) {
      v16h wf = load_frag_gf32(Wc, nt * 16, 256, kc * 32, lane);
#pragma unroll
      for (int mt = 0; mt < 4; ++mt) {
        v16h af = load_frag_lds(sCat, mt * 16, 256, kc * 32, lane);
        acc[mt] = __builtin_amdgcn_wmma_f32_16x16x32_f16(
            false, af, false, wf, (short)0, acc[mt], false, false);
      }
    }
    // fused epilogue: tanh + bc, then rank-1 contraction with Ws into energies
    const int hcol = nt * 16 + (lane & 15);
    const float wsv = Ws[hcol];
    const float bcv = bc[hcol];
#pragma unroll
    for (int mt = 0; mt < 4; ++mt) {
      const int mbase = mt * 16 + ((lane & 16) ? 8 : 0);
#pragma unroll
      for (int r = 0; r < 8; ++r) {
        float res = fast_tanh(acc[mt][r] + bcv);
        atomicAdd(&sEnergy[mbase + r], res * wsv);
      }
    }
  }
  __syncthreads();

  if (tid == 0) {
    const float bs = scale_b[user];
    float mx = -1e30f;
    for (int m = 0; m < M_; ++m) {
      float e = sEnergy[m] + bs;
      sAttn[m] = e;
      mx = fmaxf(mx, e);
    }
    float s = 0.f;
    for (int m = 0; m < M_; ++m) {
      float e = __expf(sAttn[m] - mx);
      sAttn[m] = e;
      s += e;
    }
    float inv = 1.f / s;
    for (int m = 0; m < M_; ++m) sAttn[m] *= inv;
  }
  __syncthreads();

  if (tid < M_) attn_out[(size_t)b * M_ + tid] = sAttn[tid];
  {
    float c = 0.f;
    for (int m = 0; m < M_; ++m) c += sAttn[m] * itb[m * H_ + tid];
    ctx_out[(size_t)b * H_ + tid] = c;
  }
}

// ---------------------------------------------------------------------------
// Kernel B: GRU cell (tiny; scalar f32, float4 weight streams).
// One block per batch element; weights (590 KB) are L2-resident.
__global__ __launch_bounds__(128) void gru_kernel(
    const float* __restrict__ embed, const float* __restrict__ hidden,
    const float* __restrict__ ctx, const float* __restrict__ w_ih,
    const float* __restrict__ w_hh, const float* __restrict__ b_ih,
    const float* __restrict__ b_hh, float* __restrict__ out_hnew,
    float* __restrict__ out_embed, float* __restrict__ out_gru,
    _Float16* __restrict__ hnew_f16) {
  __shared__ float sx[256];
  __shared__ float sh[128];
  const int b = blockIdx.x, t = threadIdx.x;
  sx[t] = embed[(size_t)b * E_ + t];
  sx[128 + t] = ctx[(size_t)b * H_ + t];
  sh[t] = hidden[(size_t)b * H_ + t];
  __syncthreads();

  float gi[3], gh[3];
#pragma unroll
  for (int c = 0; c < 3; ++c) {
    const int row = c * H_ + t;
    const float4* wi = (const float4*)(w_ih + (size_t)row * 256);
    float a = b_ih[row];
    for (int k = 0; k < 64; ++k) {
      float4 w4 = wi[k];
      a += sx[4 * k] * w4.x + sx[4 * k + 1] * w4.y +
           sx[4 * k + 2] * w4.z + sx[4 * k + 3] * w4.w;
    }
    gi[c] = a;
    const float4* wh = (const float4*)(w_hh + (size_t)row * 128);
    float g = b_hh[row];
    for (int k = 0; k < 32; ++k) {
      float4 w4 = wh[k];
      g += sh[4 * k] * w4.x + sh[4 * k + 1] * w4.y +
           sh[4 * k + 2] * w4.z + sh[4 * k + 3] * w4.w;
    }
    gh[c] = g;
  }
  float r = fast_sigmoid(gi[0] + gh[0]);
  float z = fast_sigmoid(gi[1] + gh[1]);
  float n = fast_tanh(gi[2] + r * gh[2]);
  float hn = (1.f - z) * n + z * sh[t];

  const size_t o = (size_t)b * H_ + t;
  out_hnew[o] = hn;
  out_gru[o] = hn;
  out_embed[o] = sx[t];
  hnew_f16[o] = (_Float16)hn;
}

// ---------------------------------------------------------------------------
// Kernel C: output = h_new @ lin_W^T + lin_b   (1024x128 @ 128x100000)
// Memory-roofline kernel (~460 MB @ 23.3 TB/s): stage a 128-column f16 strip
// of lin_W once in LDS, sweep all 1024 batch rows against it. 256 thr = 8
// waves; wave w owns one 16-col subtile and 4 M-subtiles per 64-row chunk.
__global__ __launch_bounds__(256) void proj_kernel(
    const _Float16* __restrict__ hnew, const float* __restrict__ lin_W,
    const float* __restrict__ lin_b, float* __restrict__ out) {
  __shared__ _Float16 sW[128 * 128];  // 32 KB [n][k]
  __shared__ _Float16 sA[64 * 128];   // 16 KB [m][k]
  const int tid = threadIdx.x;
  const int nbase = blockIdx.x * 128;

  for (int idx = tid; idx < 128 * 128; idx += 256) {
    int n = idx >> 7, k = idx & 127;
    int ng = nbase + n;
    sW[idx] = (ng < NI_) ? (_Float16)lin_W[(size_t)ng * H_ + k] : (_Float16)0.f;
  }

  const int wave = tid >> 5, lane = tid & 31;
  const int ncol = nbase + wave * 16 + (lane & 15);
  const float lb = (ncol < NI_) ? lin_b[ncol] : 0.f;

  for (int mc = 0; mc < B_ / 64; ++mc) {
    __syncthreads();  // W staged (mc==0) / previous chunk's reads done
    for (int idx = tid; idx < 64 * 128; idx += 256)
      sA[idx] = hnew[(size_t)(mc * 64) * H_ + idx];
    __syncthreads();

#pragma unroll
    for (int mt = 0; mt < 4; ++mt) {
      v8f acc = {};
#pragma unroll
      for (int kc = 0; kc < 4; ++kc) {
        v16h af = load_frag_lds(sA, mt * 16, 128, kc * 32, lane);
        v16h wf = load_frag_lds(sW, wave * 16, 128, kc * 32, lane);
        acc = __builtin_amdgcn_wmma_f32_16x16x32_f16(
            false, af, false, wf, (short)0, acc, false, false);
      }
      if (ncol < NI_) {
        const int mrow = mc * 64 + mt * 16 + ((lane & 16) ? 8 : 0);
#pragma unroll
        for (int r = 0; r < 8; ++r)
          out[(size_t)(mrow + r) * NI_ + ncol] = acc[r] + lb;
      }
    }
  }
}

// ---------------------------------------------------------------------------
extern "C" void kernel_launch(void* const* d_in, const int* in_sizes, int n_in,
                              void* d_out, int out_size, void* d_ws,
                              size_t ws_size, hipStream_t stream) {
  const float* embed   = (const float*)d_in[0];
  const float* hidden  = (const float*)d_in[1];
  const float* inter   = (const float*)d_in[2];
  // d_in[3] = delta_t_h, unused by the reference
  const int*   users   = (const int*)d_in[4];
  const float* w_ih    = (const float*)d_in[5];
  const float* w_hh    = (const float*)d_in[6];
  const float* b_ih    = (const float*)d_in[7];
  const float* b_hh    = (const float*)d_in[8];
  const float* lin_W   = (const float*)d_in[9];
  const float* lin_b   = (const float*)d_in[10];
  const float* cat_W   = (const float*)d_in[11];
  const float* cat_b   = (const float*)d_in[12];
  const float* scale_W = (const float*)d_in[13];
  const float* scale_b = (const float*)d_in[14];

  float* out = (float*)d_out;
  float*     ctx      = (float*)d_ws;                                    // 512 KB
  _Float16*  hnew_f16 = (_Float16*)((char*)d_ws + (size_t)B_ * H_ * 4);  // 256 KB

  attn_kernel<<<B_, 128, 0, stream>>>(hidden, inter, users, cat_W, cat_b,
                                      scale_W, scale_b, out + OUT4, ctx);

  gru_kernel<<<B_, 128, 0, stream>>>(embed, hidden, ctx, w_ih, w_hh, b_ih,
                                     b_hh, out + OUT1, out + OUT2, out + OUT3,
                                     hnew_f16);

  const int nblocks = (NI_ / 16 + 7) / 8;  // 782 strips of 128 columns
  proj_kernel<<<nblocks, 256, 0, stream>>>(hnew_f16, lin_W, lin_b, out + OUT0);
}
